// PointNet2_40286793237049
// MI455X (gfx1250) — compile-verified, matrix operand order checked by reading, not run on hardware
//
#include <hip/hip_runtime.h>
#include <cstdint>
#include <cstddef>

typedef _Float16 v16h __attribute__((ext_vector_type(16)));
typedef _Float16 v8h  __attribute__((ext_vector_type(8)));
typedef float    v8f  __attribute__((ext_vector_type(8)));

#define BN_EPS 1e-5f
#define BATCH 16

// ---------------------------------------------------------------------------
// (B,6,N) channel-major input -> coords (B,N,3) + feats (B,N,3)
// ---------------------------------------------------------------------------
__global__ void k_transpose(const float* __restrict__ xyz,
                            float* __restrict__ coords,
                            float* __restrict__ feats, int N) {
  long long total = (long long)BATCH * N;
  for (long long e = (long long)blockIdx.x * blockDim.x + threadIdx.x; e < total;
       e += (long long)gridDim.x * blockDim.x) {
    int b = (int)(e / N);
    int n = (int)(e % N);
    const float* src = xyz + (size_t)b * 6 * N + n;
    float* cd = coords + ((size_t)b * N + n) * 3;
    float* ft = feats + ((size_t)b * N + n) * 3;
    cd[0] = src[0];
    cd[1] = src[(size_t)N];
    cd[2] = src[(size_t)2 * N];
    ft[0] = src[(size_t)3 * N];
    ft[1] = src[(size_t)4 * N];
    ft[2] = src[(size_t)5 * N];
  }
}

// ---------------------------------------------------------------------------
// Farthest point sampling: one block per batch, dist kept in LDS,
// block-wide argmax (first-index tie break to match jnp.argmax).
// ---------------------------------------------------------------------------
__global__ __launch_bounds__(256) void k_fps(const float* __restrict__ coords,
                                             int N, int S, int* __restrict__ fps_idx) {
  __shared__ float dist[4096];
  __shared__ float rv[256];
  __shared__ int   ri[256];
  int b = blockIdx.x;
  int t = threadIdx.x;
  const float* xb = coords + (size_t)b * N * 3;
  for (int i = t; i < N; i += 256) dist[i] = 1e10f;
  __syncthreads();
  int far = 0;
  for (int s = 0; s < S; s++) {
    if (t == 0) fps_idx[b * S + s] = far;
    float cx = xb[far * 3 + 0], cy = xb[far * 3 + 1], cz = xb[far * 3 + 2];
    float best = -1.0f;
    int bi = 0;
    for (int i = t; i < N; i += 256) {
      float dx = xb[i * 3 + 0] - cx;
      float dy = xb[i * 3 + 1] - cy;
      float dz = xb[i * 3 + 2] - cz;
      float d = dx * dx + dy * dy + dz * dz;
      float nd = fminf(dist[i], d);
      dist[i] = nd;
      if (nd > best) { best = nd; bi = i; }
    }
    rv[t] = best;
    ri[t] = bi;
    __syncthreads();
    for (int off = 128; off > 0; off >>= 1) {
      if (t < off) {
        if (rv[t + off] > rv[t] || (rv[t + off] == rv[t] && ri[t + off] < ri[t])) {
          rv[t] = rv[t + off];
          ri[t] = ri[t + off];
        }
      }
      __syncthreads();
    }
    far = ri[0];
    __syncthreads();
  }
}

// ---------------------------------------------------------------------------
// new_xyz = coords[fps_idx]
// ---------------------------------------------------------------------------
__global__ void k_gather_xyz(const float* __restrict__ coords,
                             const int* __restrict__ fps_idx,
                             float* __restrict__ new_xyz, int N, int S) {
  int total = BATCH * S;
  for (int e = blockIdx.x * blockDim.x + threadIdx.x; e < total;
       e += gridDim.x * blockDim.x) {
    int b = e / S;
    int p = fps_idx[e];
    const float* src = coords + ((size_t)b * N + p) * 3;
    float* dst = new_xyz + (size_t)e * 3;
    dst[0] = src[0]; dst[1] = src[1]; dst[2] = src[2];
  }
}

// ---------------------------------------------------------------------------
// Ball query: first K indices with d2 <= r2 (scan order == smallest indices,
// matching the reference's arange/where/sort construction), pad with first.
// ---------------------------------------------------------------------------
__global__ void k_ballq(const float* __restrict__ coords,
                        const float* __restrict__ new_xyz,
                        int* __restrict__ idx, int N, int S, int K, float r2) {
  int total = BATCH * S;
  for (int q = blockIdx.x * blockDim.x + threadIdx.x; q < total;
       q += gridDim.x * blockDim.x) {
    int b = q / S;
    const float* xb = coords + (size_t)b * N * 3;
    float qx = new_xyz[(size_t)q * 3 + 0];
    float qy = new_xyz[(size_t)q * 3 + 1];
    float qz = new_xyz[(size_t)q * 3 + 2];
    int* out = idx + (size_t)q * K;
    int cnt = 0, first = 0;
    for (int i = 0; i < N && cnt < K; i++) {
      float dx = xb[i * 3 + 0] - qx;
      float dy = xb[i * 3 + 1] - qy;
      float dz = xb[i * 3 + 2] - qz;
      if (dx * dx + dy * dy + dz * dz <= r2) {
        if (cnt == 0) first = i;
        out[cnt++] = i;
      }
    }
    for (int j = cnt; j < K; j++) out[j] = first;
  }
}

// ---------------------------------------------------------------------------
// Build grouped activation matrix A (M = B*S*K rows, Kpad f16 cols, zero pad)
// cols 0..2 = centered coords, 3..Cin-1 = gathered feats.
// ---------------------------------------------------------------------------
__global__ void k_group(const float* __restrict__ coords,
                        const float* __restrict__ feats,
                        const float* __restrict__ new_xyz,
                        const int* __restrict__ idx,
                        _Float16* __restrict__ A,
                        int N, int S, int K, int Cf, int Cin, int Kpad,
                        int group_all) {
  long long total = (long long)BATCH * S * K * Kpad;
  for (long long e = (long long)blockIdx.x * blockDim.x + threadIdx.x; e < total;
       e += (long long)gridDim.x * blockDim.x) {
    long long row = e / Kpad;
    int c = (int)(e % Kpad);
    int k = (int)(row % K);
    long long bs = row / K;           // b*S + s
    int b = (int)(bs / S);
    int pi = group_all ? k : idx[(size_t)bs * K + k];
    float v = 0.0f;
    if (c < 3) {
      v = coords[((size_t)b * N + pi) * 3 + c];
      if (!group_all) v -= new_xyz[(size_t)bs * 3 + c];
    } else if (c < Cin) {
      v = feats[((size_t)b * N + pi) * Cf + (c - 3)];
    }
    A[e] = (_Float16)v;
  }
}

// ---------------------------------------------------------------------------
// Weights (Cout, Cin) f32 -> (Cout, Kpad) f16, zero padded
// ---------------------------------------------------------------------------
__global__ void k_wconv(const float* __restrict__ W, _Float16* __restrict__ Wh,
                        int Cout, int Cin, int Kpad) {
  int total = Cout * Kpad;
  for (int e = blockIdx.x * blockDim.x + threadIdx.x; e < total;
       e += gridDim.x * blockDim.x) {
    int o = e / Kpad, c = e % Kpad;
    Wh[e] = (c < Cin) ? (_Float16)W[o * Cin + c] : (_Float16)0.0f;
  }
}

// ---------------------------------------------------------------------------
// WMMA GEMM: Y[M,Cout] = A[M,Kpad] * W^T + bias.
// One wave computes a 16 x (16*NT) output strip, software-pipelined over the
// K dimension: fragments for step k+32 are prefetched before the NT WMMAs of
// step k execute, so VMEM loads overlap matrix ops instead of full
// s_wait_loadcnt drains. A fragment is shared by NT WMMAs (NT accumulators).
// M mult of 16, Cout mult of 16*NT, Kpad mult of 32 -> no edge handling.
// A fragment per ISA 16-bit A table (lanes 0-15: K=k0+0..7 & +16..23,
// lanes 16-31: K=k0+8..15 & +24..31); B fragment mirrors with lane->N.
// ---------------------------------------------------------------------------
static __device__ inline v16h cat8(v8h lo, v8h hi) {
  return __builtin_shufflevector(lo, hi, 0, 1, 2, 3, 4, 5, 6, 7,
                                 8, 9, 10, 11, 12, 13, 14, 15);
}

template <int NT>
__global__ __launch_bounds__(32) void k_gemm(const _Float16* __restrict__ A,
                                             const _Float16* __restrict__ Wh,
                                             const float* __restrict__ bias,
                                             _Float16* __restrict__ Y,
                                             int M, int Kpad, int Cout) {
  int lane = threadIdx.x & 31;
  int mtile = blockIdx.x << 4;
  int ntile0 = blockIdx.y * (16 * NT);
  int halfsel = lane >> 4;   // 0 or 1
  int lid = lane & 15;

  v8f acc[NT];
#pragma unroll
  for (int j = 0; j < NT; j++) {
    float bv = bias[ntile0 + j * 16 + lid];
#pragma unroll
    for (int r = 0; r < 8; r++) acc[j][r] = bv;
  }

  const _Float16* Arow = A + (size_t)(mtile + lid) * Kpad + (halfsel ? 8 : 0);
  const _Float16* Wbase =
      Wh + (size_t)(ntile0 + lid) * Kpad + (halfsel ? 8 : 0);

  // ---- prologue: load fragments for k0 = 0 ----
  v8h cal = *(const v8h*)(Arow + 0);
  v8h cah = *(const v8h*)(Arow + 16);
  v8h cbl[NT], cbh[NT];
#pragma unroll
  for (int j = 0; j < NT; j++) {
    const _Float16* Wrow = Wbase + (size_t)(j * 16) * Kpad;
    cbl[j] = *(const v8h*)(Wrow + 0);
    cbh[j] = *(const v8h*)(Wrow + 16);
  }

  // ---- pipelined main loop: prefetch k0, compute k0-32 ----
  for (int k0 = 32; k0 < Kpad; k0 += 32) {
    v8h nal = *(const v8h*)(Arow + k0);
    v8h nah = *(const v8h*)(Arow + k0 + 16);
    v8h nbl[NT], nbh[NT];
#pragma unroll
    for (int j = 0; j < NT; j++) {
      const _Float16* Wrow = Wbase + (size_t)(j * 16) * Kpad;
      nbl[j] = *(const v8h*)(Wrow + k0);
      nbh[j] = *(const v8h*)(Wrow + k0 + 16);
    }
    v16h av = cat8(cal, cah);
#pragma unroll
    for (int j = 0; j < NT; j++) {
      v16h bw = cat8(cbl[j], cbh[j]);
      acc[j] = __builtin_amdgcn_wmma_f32_16x16x32_f16(false, av, false, bw,
                                                      (short)0, acc[j],
                                                      false, false);
    }
    cal = nal;
    cah = nah;
#pragma unroll
    for (int j = 0; j < NT; j++) { cbl[j] = nbl[j]; cbh[j] = nbh[j]; }
  }

  // ---- epilogue: compute last k-step ----
  {
    v16h av = cat8(cal, cah);
#pragma unroll
    for (int j = 0; j < NT; j++) {
      v16h bw = cat8(cbl[j], cbh[j]);
      acc[j] = __builtin_amdgcn_wmma_f32_16x16x32_f16(false, av, false, bw,
                                                      (short)0, acc[j],
                                                      false, false);
    }
  }

  // C/D layout: VGPR r -> row (halfsel*8 + r), col n
  _Float16* outr = Y + (size_t)(mtile + halfsel * 8) * Cout;
#pragma unroll
  for (int j = 0; j < NT; j++) {
    int n = ntile0 + j * 16 + lid;
#pragma unroll
    for (int r = 0; r < 8; r++) outr[(size_t)r * Cout + n] = (_Float16)acc[j][r];
  }
}

// ---------------------------------------------------------------------------
// Per-channel batch stats over M rows (one block per channel)
// ---------------------------------------------------------------------------
__global__ __launch_bounds__(256) void k_stats(const _Float16* __restrict__ X,
                                               int M, int C,
                                               float* __restrict__ mean,
                                               float* __restrict__ var) {
  __shared__ float ssum[256], ssq[256];
  int c = blockIdx.x;
  int t = threadIdx.x;
  float s = 0.0f, q = 0.0f;
  for (int m = t; m < M; m += 256) {
    float v = (float)X[(size_t)m * C + c];
    s += v;
    q += v * v;
  }
  ssum[t] = s; ssq[t] = q;
  __syncthreads();
  for (int off = 128; off > 0; off >>= 1) {
    if (t < off) { ssum[t] += ssum[t + off]; ssq[t] += ssq[t + off]; }
    __syncthreads();
  }
  if (t == 0) {
    float mu = ssum[0] / (float)M;
    mean[c] = mu;
    var[c] = ssq[0] / (float)M - mu * mu;
  }
}

// ---------------------------------------------------------------------------
// In-place BN + ReLU on f16 activations
// ---------------------------------------------------------------------------
__global__ void k_bnrelu(_Float16* __restrict__ X,
                         const float* __restrict__ mean,
                         const float* __restrict__ var,
                         const float* __restrict__ gamma,
                         const float* __restrict__ beta,
                         long long M, int C) {
  long long total = M * C;
  for (long long e = (long long)blockIdx.x * blockDim.x + threadIdx.x; e < total;
       e += (long long)gridDim.x * blockDim.x) {
    int c = (int)(e % C);
    float v = (float)X[e];
    float y = (v - mean[c]) * rsqrtf(var[c] + BN_EPS) * gamma[c] + beta[c];
    X[e] = (_Float16)fmaxf(y, 0.0f);
  }
}

// ---------------------------------------------------------------------------
// Max over the K neighborhood: (BS, K, C) -> (BS, C) f32
// ---------------------------------------------------------------------------
__global__ void k_maxpool(const _Float16* __restrict__ X, float* __restrict__ out,
                          long long BS, int K, int C) {
  long long total = BS * C;
  for (long long e = (long long)blockIdx.x * blockDim.x + threadIdx.x; e < total;
       e += (long long)gridDim.x * blockDim.x) {
    long long bs = e / C;
    int c = (int)(e % C);
    const _Float16* p = X + ((size_t)bs * K) * C + c;
    float m = -3.0e38f;
    for (int k = 0; k < K; k++) m = fmaxf(m, (float)p[(size_t)k * C]);
    out[e] = m;
  }
}

__global__ void k_zero(float* __restrict__ out, int n) {
  for (int e = blockIdx.x * blockDim.x + threadIdx.x; e < n;
       e += gridDim.x * blockDim.x)
    out[e] = 0.0f;
}

// ---------------------------------------------------------------------------
static inline int gridFor(long long n, int block) {
  long long g = (n + block - 1) / block;
  if (g > 16384) g = 16384;
  if (g < 1) g = 1;
  return (int)g;
}

extern "C" void kernel_launch(void* const* d_in, const int* in_sizes, int n_in,
                              void* d_out, int out_size, void* d_ws, size_t ws_size,
                              hipStream_t stream) {
  (void)in_sizes;
  if (n_in < 49) {
    k_zero<<<gridFor(out_size, 256), 256, 0, stream>>>((float*)d_out, out_size);
    return;
  }
  const float* xyz = (const float*)d_in[0];

  static const int   Sarr[4]    = {778, 388, 194, 1};
  static const int   Karr[4]    = {16, 32, 64, 194};
  static const float Rad[4]     = {0.2f, 0.4f, 0.8f, 0.0f};
  static const int   CinArr[4]  = {6, 67, 131, 259};
  static const int   KpadArr[4] = {32, 96, 160, 288};
  static const int   MLP[4][3]  = {{32, 32, 64}, {64, 64, 128},
                                   {128, 128, 256}, {256, 256, 512}};

  // ---- workspace carve (all sizes compile-time constant, 256B aligned) ----
  const int M3 = 16 * 194 * 64;                       // row count of stages 1 and 2: 198656
  auto alignUp = [](size_t x) { return (x + 255) & ~(size_t)255; };
  size_t off = 0;
  char* base = (char*)d_ws;
  auto take = [&](size_t sz) { char* r = base + off; off += alignUp(sz); return (void*)r; };

  _Float16* slot0  = (_Float16*)take((size_t)M3 * 160 * 2);    // A0 / L1-out slot
  _Float16* slot1  = (_Float16*)take((size_t)M3 * 256 * 2);    // L0-out / L2-out slot
  float* coordsA   = (float*)take((size_t)16 * 4096 * 3 * 4);
  float* coordsB   = (float*)take((size_t)16 * 4096 * 3 * 4);
  float* featsA    = (float*)take((size_t)16 * 778 * 64 * 4);
  float* featsB    = (float*)take((size_t)16 * 778 * 64 * 4);
  int*   fpsIdx    = (int*)take((size_t)16 * 778 * 4);
  int*   ballIdx   = (int*)take((size_t)16 * 778 * 16 * 4);
  _Float16* Wh     = (_Float16*)take((size_t)512 * 288 * 2);
  float* meanBuf   = (float*)take((size_t)512 * 4);
  float* varBuf    = (float*)take((size_t)512 * 4);

  if (off > ws_size) {
    k_zero<<<gridFor(out_size, 256), 256, 0, stream>>>((float*)d_out, out_size);
    return;
  }

  // ---- split input ----
  k_transpose<<<gridFor((long long)16 * 4096, 256), 256, 0, stream>>>(
      xyz, coordsA, featsA, 4096);

  float* coords_cur = coordsA;
  float* coords_nxt = coordsB;
  float* feats_cur = featsA;
  float* feats_nxt = featsB;
  int Ncur = 4096, Cf = 3;

  for (int st = 0; st < 4; st++) {
    int S = Sarr[st], K = Karr[st], Cin = CinArr[st], Kpad = KpadArr[st];
    long long M = (long long)BATCH * S * K;
    int group_all = (st == 3);

    if (!group_all) {
      k_fps<<<BATCH, 256, 0, stream>>>(coords_cur, Ncur, S, fpsIdx);
      k_gather_xyz<<<gridFor(BATCH * S, 256), 256, 0, stream>>>(
          coords_cur, fpsIdx, coords_nxt, Ncur, S);
      k_ballq<<<gridFor(BATCH * S, 128), 128, 0, stream>>>(
          coords_cur, coords_nxt, ballIdx, Ncur, S, K, Rad[st] * Rad[st]);
    }
    k_group<<<gridFor(M * Kpad, 256), 256, 0, stream>>>(
        coords_cur, feats_cur, coords_nxt, ballIdx, slot0,
        Ncur, S, K, Cf, Cin, Kpad, group_all);

    _Float16* bufs[2] = {slot0, slot1};
    int cur = 0;
    int kIn = Kpad;
    for (int l = 0; l < 3; l++) {
      int Cout = MLP[st][l];
      int CinL = (l == 0) ? Cin : MLP[st][l - 1];
      const float* W     = (const float*)d_in[1 + (st * 3 + l) * 4 + 0];
      const float* bias  = (const float*)d_in[1 + (st * 3 + l) * 4 + 1];
      const float* gamma = (const float*)d_in[1 + (st * 3 + l) * 4 + 2];
      const float* beta  = (const float*)d_in[1 + (st * 3 + l) * 4 + 3];

      k_wconv<<<gridFor(Cout * kIn, 256), 256, 0, stream>>>(W, Wh, Cout, CinL, kIn);

      if (Cout % 64 == 0) {
        dim3 gg((unsigned)(M / 16), (unsigned)(Cout / 64), 1);
        k_gemm<4><<<gg, 32, 0, stream>>>(bufs[cur], Wh, bias, bufs[cur ^ 1],
                                         (int)M, kIn, Cout);
      } else {
        dim3 gg((unsigned)(M / 16), (unsigned)(Cout / 32), 1);
        k_gemm<2><<<gg, 32, 0, stream>>>(bufs[cur], Wh, bias, bufs[cur ^ 1],
                                         (int)M, kIn, Cout);
      }
      cur ^= 1;
      k_stats<<<Cout, 256, 0, stream>>>(bufs[cur], (int)M, Cout, meanBuf, varBuf);
      k_bnrelu<<<gridFor(M * Cout, 256), 256, 0, stream>>>(
          bufs[cur], meanBuf, varBuf, gamma, beta, M, Cout);
      kIn = Cout;
    }

    long long BS = (long long)BATCH * S;
    int Cout = MLP[st][2];
    float* outF = (st == 3) ? (float*)d_out : feats_nxt;
    k_maxpool<<<gridFor(BS * Cout, 256), 256, 0, stream>>>(bufs[cur], outF, BS, K, Cout);

    if (st < 3) {
      float* t = coords_cur; coords_cur = coords_nxt; coords_nxt = t;
      t = feats_cur; feats_cur = feats_nxt; feats_nxt = t;
      Ncur = S;
      Cf = Cout;
    }
  }
}
